// SIGNET_static_62251255988445
// MI455X (gfx1250) — matrix operand order repaired and checked by reading, not compiled, not run on hardware
//
#include <hip/hip_runtime.h>

typedef __attribute__((ext_vector_type(16))) _Float16 v16h;
typedef __attribute__((ext_vector_type(8)))  _Float16 v8h;
typedef __attribute__((ext_vector_type(8)))  float    v8f;

#define MT     64          // rows per workgroup (4 M-tiles of 16)
#define PITCH  520         // halfs per LDS row (512 + 8 pad -> conflict-free strides)

struct SignetParams {
  const float*    x;        // [n,4]
  const _Float16* wh;       // f16 weights, layers 0..8, packed sequentially
  const float*    bias[9];  // f32 biases, layers 0..8
  const float*    w10;      // [3,512] f32
  const float*    b10;      // [3] f32
  float*          out;      // [n,3]
  int             nrows;
};

// ---------------------------------------------------------------------------
// weight f32 -> f16 conversion (runs once before the fused kernel)
// ---------------------------------------------------------------------------
__global__ void signet_cvt_w(const float* __restrict__ src,
                             _Float16* __restrict__ dst, int n) {
  for (int i = blockIdx.x * blockDim.x + threadIdx.x; i < n;
       i += gridDim.x * blockDim.x)
    dst[i] = (_Float16)src[i];
}

// ---------------------------------------------------------------------------
// fully fused SIGNET forward: embed -> 9 sine layers (WMMA) -> linear(512,3)
// 256 threads = 8 waves; each wave computes a 64-row x 64-col output slab
// as 4x4 WMMA tiles, so every A and B fragment is reused 4x per load.
// ---------------------------------------------------------------------------
__global__ __launch_bounds__(256)
void signet_fused(SignetParams p) {
  extern __shared__ __align__(16) char smem_raw[];
  _Float16* E  = (_Float16*)smem_raw;       // embedding (persists for skip)
  _Float16* S0 = E  + MT * PITCH;           // ping
  _Float16* S1 = S0 + MT * PITCH;           // pong

  const int tid  = threadIdx.x;
  const int row0 = blockIdx.x * MT;

  // ---- Gegenbauer(alpha=0.5)=Legendre features: 64 rows x 4 coords = 256 thr
  {
    const int r = tid >> 2, c = tid & 3;
    const int grow = row0 + r;
    const float xv = (grow < p.nrows) ? p.x[grow * 4 + c] : 0.0f;
    const int N    = (c < 2) ? 16 : 240;
    const int base = (c == 0) ? 0 : (c == 1) ? 16 : (c == 2) ? 32 : 272;
    _Float16* dst = E + r * PITCH + base;
    float cm2 = 1.0f, cm1 = xv;              // c1 = 2*alpha*x = x
    dst[0] = (_Float16)cm1;
    for (int i = 2; i <= N; ++i) {
      float ci = ((2.0f * (float)i - 1.0f) * xv * cm1 -
                  ((float)i - 1.0f) * cm2) / (float)i;
      dst[i - 1] = (_Float16)ci;
      cm2 = cm1; cm1 = ci;
    }
  }
  __syncthreads();

  const int lane  = tid & 31;
  const int wv    = tid >> 5;       // wave 0..7
  const int nlane = lane & 15;      // M-row for A-frag / N-col for B-frag
  const int half  = lane >> 4;      // lane-half per ISA fragment layouts
  const int n0    = wv * 64;        // each wave owns 64 output columns

  const _Float16* X = E;            // layer-0 input is the embedding
  int pp = 0;
  size_t woff = 0;

  union AV { v16h v; v8h h[2]; };

  for (int layer = 0; layer < 9; ++layer) {
    const int Kin = (layer == 3) ? 1024 : 512;   // skip-concat layer is 1024 wide
    const _Float16* W = p.wh + woff;
    _Float16* Y = pp ? S1 : S0;

    v8f acc[4][4];                  // [m-tile][n-tile]
#pragma unroll
    for (int mt = 0; mt < 4; ++mt)
#pragma unroll
      for (int nt = 0; nt < 4; ++nt)
#pragma unroll
        for (int r = 0; r < 8; ++r) acc[mt][nt][r] = 0.0f;

    const int nkc = Kin >> 5;
    for (int kc = 0; kc < nkc; ++kc) {
      // skip layer: logical input is [emb | act]; chunks 0..15 -> E, 16.. -> X
      const _Float16* Xs = X;
      int kcol = kc << 5;
      if (layer == 3) {
        if (kc < 16) { Xs = E; }
        else         { Xs = X; kcol = (kc - 16) << 5; }
      }

      // 4 A fragments (16x32 f16) from LDS:
      // lane holds K = half*16..+16 (contiguous) of row (mt*16 + nlane)
      AV a[4];
#pragma unroll
      for (int mt = 0; mt < 4; ++mt) {
        const _Float16* ap = Xs + (mt * 16 + nlane) * PITCH + kcol + half * 16;
        a[mt].h[0] = *(const v8h*)(ap);
        a[mt].h[1] = *(const v8h*)(ap + 8);
      }

      // 4 B fragments (32x16 f16) from L2-resident weights:
      // lane n holds 16 contiguous K-halfs of W row (n0 + nt*16 + n)
      const int kw = (kc << 5) + half * 16;
      v16h b[4];
#pragma unroll
      for (int nt = 0; nt < 4; ++nt) {
        const _Float16* bp = W + (size_t)(n0 + nt * 16 + nlane) * Kin + kw;
        b[nt] = *(const v16h*)bp;
      }

      // 16 independent WMMA chains
#pragma unroll
      for (int mt = 0; mt < 4; ++mt)
#pragma unroll
        for (int nt = 0; nt < 4; ++nt)
          acc[mt][nt] = __builtin_amdgcn_wmma_f32_16x16x32_f16(
              false, a[mt].v, false, b[nt], (short)0, acc[mt][nt], false, false);
    }

    // epilogue: bias + sin(30*x), f32 acc -> f16 activation in LDS
    const float* bv = p.bias[layer];
#pragma unroll
    for (int nt = 0; nt < 4; ++nt) {
      const int col = n0 + nt * 16 + nlane;
      const float b = bv[col];
#pragma unroll
      for (int mt = 0; mt < 4; ++mt) {
#pragma unroll
        for (int r = 0; r < 8; ++r) {
          // C/D layout: VGPR r -> row (mt*16 + half*8 + r), col = lane%16
          float v = __sinf(30.0f * (acc[mt][nt][r] + b));
          Y[(mt * 16 + half * 8 + r) * PITCH + col] = (_Float16)v;
        }
      }
    }
    __syncthreads();

    X = Y; pp ^= 1;
    woff += (size_t)512 * Kin;
  }

  // ---- final Linear(512,3), pure f32 (negligible work): 64 rows x 3 cols
  if (tid < MT * 3) {
    const int r = tid / 3, j = tid % 3;
    const int grow = row0 + r;
    if (grow < p.nrows) {
      const _Float16* xr = X + r * PITCH;
      const float*    wr = p.w10 + j * 512;
      float s = p.b10[j];
#pragma unroll 8
      for (int k = 0; k < 512; ++k) s += (float)xr[k] * wr[k];
      p.out[grow * 3 + j] = s;
    }
  }
}

// ---------------------------------------------------------------------------
extern "C" void kernel_launch(void* const* d_in, const int* in_sizes, int n_in,
                              void* d_out, int out_size, void* d_ws, size_t ws_size,
                              hipStream_t stream) {
  (void)n_in; (void)out_size; (void)ws_size;
  const int nrows = in_sizes[0] / 4;

  // convert w1..w9 (f32) -> f16 into d_ws, packed sequentially
  _Float16* wh = (_Float16*)d_ws;
  size_t off = 0;
  for (int i = 0; i < 9; ++i) {
    const int Kin = (i == 3) ? 1024 : 512;
    const int cnt = 512 * Kin;
    const float* src = (const float*)d_in[1 + 2 * i];
    int grid = (cnt + 255) / 256;
    if (grid > 1024) grid = 1024;
    signet_cvt_w<<<grid, 256, 0, stream>>>(src, wh + off, cnt);
    off += (size_t)cnt;
  }

  SignetParams p;
  p.x  = (const float*)d_in[0];
  p.wh = wh;
  for (int i = 0; i < 9; ++i) p.bias[i] = (const float*)d_in[2 + 2 * i];
  p.w10 = (const float*)d_in[19];
  p.b10 = (const float*)d_in[20];
  p.out = (float*)d_out;
  p.nrows = nrows;

  const size_t shbytes = (size_t)3 * MT * PITCH * sizeof(_Float16); // ~195 KB
  (void)hipFuncSetAttribute((const void*)signet_fused,
                            hipFuncAttributeMaxDynamicSharedMemorySize,
                            (int)shbytes);
  signet_fused<<<(nrows + MT - 1) / MT, 256, shbytes, stream>>>(p);
}
